// AniFlowFormerT_17738214932880
// MI455X (gfx1250) — matrix operand order repaired and checked by reading, not compiled, not run on hardware
//
#include <hip/hip_runtime.h>
#include <math.h>

typedef __attribute__((ext_vector_type(16))) _Float16 v16h;
typedef __attribute__((ext_vector_type(8)))  _Float16 v8h;
typedef __attribute__((ext_vector_type(8)))  float    v8f;
typedef __attribute__((ext_vector_type(4)))  int      v4i;

#define NTOK   1024
#define DMODEL 192
#define NHEAD  4
#define HDIM   48

// gfx1250 async global->LDS (ASYNCcnt) if the toolchain exposes it.
#if defined(__has_builtin)
# if __has_builtin(__builtin_amdgcn_global_load_async_to_lds_b128)
#  define HAVE_ASYNC_LDS 1
# endif
#endif
#ifndef HAVE_ASYNC_LDS
# define HAVE_ASYNC_LDS 0
#endif

__device__ __forceinline__ float4 ld4(const float* p) { return *(const float4*)p; }

#if HAVE_ASYNC_LDS
// 16-byte async copy global -> LDS (no VGPR round trip; tracked by ASYNCcnt).
__device__ __forceinline__ void async_ld16(const float* g, float* l) {
    __builtin_amdgcn_global_load_async_to_lds_b128(
        (__attribute__((address_space(1))) v4i*)g,
        (__attribute__((address_space(3))) v4i*)l, 0, 0);
}
#endif

__device__ __forceinline__ v16h cat8(v8h a, v8h b) {
    return __builtin_shufflevector(a, b, 0, 1, 2, 3, 4, 5, 6, 7,
                                   8, 9, 10, 11, 12, 13, 14, 15);
}

// pack 16 floats (4x float4) * s into a v16h fragment
__device__ __forceinline__ v16h pack16s(float4 a, float4 b, float4 c, float4 d, float s) {
    v16h r;
    r[0]  = (_Float16)(a.x * s); r[1]  = (_Float16)(a.y * s);
    r[2]  = (_Float16)(a.z * s); r[3]  = (_Float16)(a.w * s);
    r[4]  = (_Float16)(b.x * s); r[5]  = (_Float16)(b.y * s);
    r[6]  = (_Float16)(b.z * s); r[7]  = (_Float16)(b.w * s);
    r[8]  = (_Float16)(c.x * s); r[9]  = (_Float16)(c.y * s);
    r[10] = (_Float16)(c.z * s); r[11] = (_Float16)(c.w * s);
    r[12] = (_Float16)(d.x * s); r[13] = (_Float16)(d.y * s);
    r[14] = (_Float16)(d.z * s); r[15] = (_Float16)(d.w * s);
    return r;
}

// ---------------------------------------------------------------------------
// Fused local correlation (R=3 -> 49 shifts, C=128, 32x32) + token projection.
// ---------------------------------------------------------------------------
__global__ __launch_bounds__(192) void corr_tok_kernel(
    const float* __restrict__ f1, const float* __restrict__ f2,
    const float* __restrict__ tw, const float* __restrict__ tb,
    float* __restrict__ tok)
{
    __shared__ float f1s[128];
    __shared__ float corr[49];
    int n = blockIdx.x;
    int y = n >> 5, x = n & 31;
    int t = threadIdx.x;
    if (t < 128) f1s[t] = f1[t * 1024 + n];
    __syncthreads();
    if (t < 49) {
        int dy = t / 7 - 3, dx = t % 7 - 3;
        int ys = (y - dy) & 31, xs = (x - dx) & 31;
        const float* p = f2 + ys * 32 + xs;
        float s = 0.f;
        #pragma unroll 4
        for (int c = 0; c < 128; ++c) s += f1s[c] * p[c * 1024];
        corr[t] = s * 0.08838834764831845f;  // 1/sqrt(128)
    }
    __syncthreads();
    float s = tb[t];
    #pragma unroll
    for (int d = 0; d < 49; ++d) s += tw[t * 49 + d] * corr[d];
    tok[n * DMODEL + t] = s;
}

// ---------------------------------------------------------------------------
// LayerNorm over last dim (192). One block per token row.
// ---------------------------------------------------------------------------
__global__ __launch_bounds__(192) void layernorm_kernel(
    const float* __restrict__ x, const float* __restrict__ w,
    const float* __restrict__ b, float* __restrict__ y)
{
    __shared__ float red[192];
    int n = blockIdx.x, t = threadIdx.x;
    float v = x[n * DMODEL + t];
    red[t] = v;
    __syncthreads();
    for (int s = 96; s >= 3; s >>= 1) {
        if (t < s) red[t] += red[t + s];
        __syncthreads();
    }
    float mean = (red[0] + red[1] + red[2]) * (1.0f / 192.0f);
    __syncthreads();
    float d = v - mean;
    red[t] = d * d;
    __syncthreads();
    for (int s = 96; s >= 3; s >>= 1) {
        if (t < s) red[t] += red[t + s];
        __syncthreads();
    }
    float var = (red[0] + red[1] + red[2]) * (1.0f / 192.0f);
    float inv = rsqrtf(var + 1e-5f);
    y[n * DMODEL + t] = d * inv * w[t] + b[t];
}

// ---------------------------------------------------------------------------
// WMMA GEMM: Y[M,N] = f(X[M,K] @ W[N,K]^T + bias) (+ residual R)
// op: 0 = none, 1 = exact GELU, 2 = residual add.
// REQUIRES: M % 32 == 0, K % 32 == 0 (true at all call sites). N may be ragged.
// Double-buffered fp32 tiles in LDS, filled via async global->LDS (ASYNCcnt)
// when available; next tile's fill overlaps the current tile's WMMA; one
// barrier per K-step. Fragments read as b128 and converted at build time.
// ---------------------------------------------------------------------------
__global__ __launch_bounds__(128) void gemm_wmma_kernel(
    const float* __restrict__ X, const float* __restrict__ W,
    const float* __restrict__ bias, const float* __restrict__ R,
    float* __restrict__ Y, int M, int K, int N, int op)
{
    __shared__ float As[2][32][36];   // [buf][m][k], rows padded to 144 B
    __shared__ float Bs[2][32][36];   // [buf][n][k]  (W is (N,K) row-major)
    int bm = blockIdx.x * 32;
    int bn = blockIdx.y * 32;
    int tid = threadIdx.x;
    int lane = tid & 31;
    int wave = tid >> 5;
    int wr = wave >> 1, wc = wave & 1;
    int l16 = (lane >> 4) & 1;
    int ln  = lane & 15;

    int fr = tid >> 3;            // fill rows: fr and fr+16
    int fc = (tid & 7) << 2;      // fill column (float quad)

    auto fill = [&](int buf, int k0) {
        #pragma unroll
        for (int it = 0; it < 2; ++it) {
            int r = fr + it * 16;
            const float* ga = &X[(size_t)(bm + r) * K + k0 + fc];
            float* la = &As[buf][r][fc];
            int gn = bn + r;
            float* lb = &Bs[buf][r][fc];
#if HAVE_ASYNC_LDS
            async_ld16(ga, la);
            if (gn < N) {
                async_ld16(&W[(size_t)gn * K + k0 + fc], lb);
            } else {
                *(float4*)lb = make_float4(0.f, 0.f, 0.f, 0.f);
            }
#else
            *(float4*)la = ld4(ga);
            float4 v = make_float4(0.f, 0.f, 0.f, 0.f);
            if (gn < N) v = ld4(&W[(size_t)gn * K + k0 + fc]);
            *(float4*)lb = v;
#endif
        }
    };

    v8f acc = {};
    int steps = K >> 5;
    fill(0, 0);
    for (int s = 0; s < steps; ++s) {
        int buf = s & 1;
#if HAVE_ASYNC_LDS
        asm volatile("s_wait_asynccnt 0x0" ::: "memory");
#endif
        __syncthreads();
        if (s + 1 < steps) fill(1 - buf, (s + 1) << 5);

        const float* ap = As[buf][wr * 16 + ln];
        int oa = l16 << 3;
        v16h av = pack16s(*(const float4*)&ap[oa], *(const float4*)&ap[oa + 4],
                          *(const float4*)&ap[16 + oa], *(const float4*)&ap[16 + oa + 4],
                          1.f);
        const float* bp = Bs[buf][wc * 16 + ln];
        int ob = l16 << 3;
        v16h bv = pack16s(*(const float4*)&bp[ob], *(const float4*)&bp[ob + 4],
                          *(const float4*)&bp[16 + ob], *(const float4*)&bp[16 + ob + 4],
                          1.f);
        acc = __builtin_amdgcn_wmma_f32_16x16x32_f16(false, av, false, bv,
                                                     (short)0, acc, false, false);
    }
    #pragma unroll
    for (int r = 0; r < 8; ++r) {
        int gm = bm + wr * 16 + r + (l16 << 3);
        int gn = bn + wc * 16 + ln;
        if (gn < N) {
            float v = acc[r] + bias[gn];
            if (op == 1) v = 0.5f * v * (1.f + erff(v * 0.70710678118654752f));
            else if (op == 2) v += R[(size_t)gm * N + gn];
            Y[(size_t)gm * N + gn] = v;
        }
    }
}

// ---------------------------------------------------------------------------
// Fused flash attention. qkv: (N,576) rows = [q(192) k(192) v(192)].
// grid = (N/16 query tiles, 4 heads), block = 32 (one wave).
// HD=48 zero-padded to 64 for QK^T; key blocks of 32; online softmax.
// ---------------------------------------------------------------------------
__global__ __launch_bounds__(32) void attn_kernel(
    const float* __restrict__ qkv, float* __restrict__ out, int Nt)
{
    int qt = blockIdx.x, h = blockIdx.y;
    int lane = threadIdx.x;
    int l16 = lane >> 4, ln = lane & 15;
    int m0 = qt * 16;
    const float scale = 0.14433756729740643f;  // 1/sqrt(48)
    const float4 z4 = make_float4(0.f, 0.f, 0.f, 0.f);

    const float* qrow = qkv + (size_t)(m0 + ln) * 576 + h * HDIM;
    int oa = l16 << 4;  // 0 or 16
    v16h qa0 = pack16s(ld4(qrow + oa), ld4(qrow + oa + 4),
                       ld4(qrow + oa + 8), ld4(qrow + oa + 12), scale);
    v16h qa1 = pack16s(ld4(qrow + 32), ld4(qrow + 36), ld4(qrow + 40), ld4(qrow + 44), l16 ? 0.f : scale);

    v8f o0 = {}, o1 = {}, o2 = {};
    float mrow[8], lrow[8];
    #pragma unroll
    for (int r = 0; r < 8; ++r) { mrow[r] = -1e30f; lrow[r] = 0.f; }

    __shared__ _Float16 P[16][40];    // probabilities, row-padded
    __shared__ _Float16 Vt[48][40];   // V tile transposed: [vdim][key]

    for (int kb = 0; kb < Nt; kb += 32) {
        const float* kr0 = qkv + (size_t)(kb + ln) * 576 + 192 + h * HDIM;
        const float* kr1 = kr0 + 16 * 576;
        int ob = l16 << 4;  // 0 or 16
        v16h k0a = pack16s(ld4(kr0 + ob), ld4(kr0 + ob + 4),
                           ld4(kr0 + ob + 8), ld4(kr0 + ob + 12), 1.f);
        v16h k1a = pack16s(ld4(kr1 + ob), ld4(kr1 + ob + 4),
                           ld4(kr1 + ob + 8), ld4(kr1 + ob + 12), 1.f);
        float4 c0 = z4, c1 = z4, c2 = z4, c3 = z4, d0 = z4, d1 = z4, d2 = z4, d3 = z4;
        if (!l16) {  // k-dims 48..63 are padding -> zero for upper half-lanes
            c0 = ld4(kr0 + 32); c1 = ld4(kr0 + 36); c2 = ld4(kr0 + 40); c3 = ld4(kr0 + 44);
            d0 = ld4(kr1 + 32); d1 = ld4(kr1 + 36); d2 = ld4(kr1 + 40); d3 = ld4(kr1 + 44);
        }
        v16h k0b = pack16s(c0, c1, c2, c3, 1.f);
        v16h k1b = pack16s(d0, d1, d2, d3, 1.f);

        v8f s0 = {}, s1 = {};
        s0 = __builtin_amdgcn_wmma_f32_16x16x32_f16(false, qa0, false, k0a, (short)0, s0, false, false);
        s0 = __builtin_amdgcn_wmma_f32_16x16x32_f16(false, qa1, false, k0b, (short)0, s0, false, false);
        s1 = __builtin_amdgcn_wmma_f32_16x16x32_f16(false, qa0, false, k1a, (short)0, s1, false, false);
        s1 = __builtin_amdgcn_wmma_f32_16x16x32_f16(false, qa1, false, k1b, (short)0, s1, false, false);

        // stage V tile (32 keys x 48 dims) transposed into LDS
        const float* vrow = qkv + (size_t)(kb + lane) * 576 + 384 + h * HDIM;
        #pragma unroll
        for (int j = 0; j < 12; ++j) {
            float4 v = ld4(vrow + 4 * j);
            Vt[4 * j + 0][lane] = (_Float16)v.x;
            Vt[4 * j + 1][lane] = (_Float16)v.y;
            Vt[4 * j + 2][lane] = (_Float16)v.z;
            Vt[4 * j + 3][lane] = (_Float16)v.w;
        }

        // online softmax
        #pragma unroll
        for (int r = 0; r < 8; ++r) {
            float v = fmaxf(s0[r], s1[r]);
            #pragma unroll
            for (int mask = 1; mask <= 8; mask <<= 1)
                v = fmaxf(v, __shfl_xor(v, mask, 32));
            float mnew = fmaxf(mrow[r], v);
            float alpha = __expf(mrow[r] - mnew);
            float p0 = __expf(s0[r] - mnew);
            float p1 = __expf(s1[r] - mnew);
            float rs = p0 + p1;
            #pragma unroll
            for (int mask = 1; mask <= 8; mask <<= 1)
                rs += __shfl_xor(rs, mask, 32);
            lrow[r] = lrow[r] * alpha + rs;
            mrow[r] = mnew;
            o0[r] *= alpha; o1[r] *= alpha; o2[r] *= alpha;
            P[r + 8 * l16][ln]      = (_Float16)p0;
            P[r + 8 * l16][16 + ln] = (_Float16)p1;
        }
        __syncthreads();

        // P @ V
        v8h p0v = *(const v8h*)&P[ln][l16 << 3];
        v8h p1v = *(const v8h*)&P[ln][16 + (l16 << 3)];
        v16h pa = cat8(p0v, p1v);
        #pragma unroll
        for (int t = 0; t < 3; ++t) {
            int vr = t * 16 + ln;
            v8h vb0 = *(const v8h*)&Vt[vr][l16 << 3];
            v8h vb1 = *(const v8h*)&Vt[vr][16 + (l16 << 3)];
            v16h vb = cat8(vb0, vb1);
            if (t == 0)
                o0 = __builtin_amdgcn_wmma_f32_16x16x32_f16(false, pa, false, vb, (short)0, o0, false, false);
            else if (t == 1)
                o1 = __builtin_amdgcn_wmma_f32_16x16x32_f16(false, pa, false, vb, (short)0, o1, false, false);
            else
                o2 = __builtin_amdgcn_wmma_f32_16x16x32_f16(false, pa, false, vb, (short)0, o2, false, false);
        }
        __syncthreads();
    }
    #pragma unroll
    for (int r = 0; r < 8; ++r) {
        float inv = 1.f / lrow[r];
        int gm = m0 + r + 8 * l16;
        out[(size_t)gm * DMODEL + h * HDIM + ln]      = o0[r] * inv;
        out[(size_t)gm * DMODEL + h * HDIM + 16 + ln] = o1[r] * inv;
        out[(size_t)gm * DMODEL + h * HDIM + 32 + ln] = o2[r] * inv;
    }
}

// ---------------------------------------------------------------------------
// Elementwise EMA blend: X = 0.8*X + 0.2*S
// ---------------------------------------------------------------------------
__global__ void blend_kernel(float* __restrict__ X, const float* __restrict__ S, int n)
{
    int i = blockIdx.x * blockDim.x + threadIdx.x;
    if (i < n) X[i] = 0.8f * X[i] + 0.2f * S[i];
}

// (N,2) token-major -> (2,N) planar flow
__global__ void nc_to_cn_kernel(const float* __restrict__ in, float* __restrict__ out, int n)
{
    int i = blockIdx.x * blockDim.x + threadIdx.x;
    if (i < 2 * n) {
        int c = i / n, p = i - c * n;
        out[i] = in[p * 2 + c];
    }
}

// ---------------------------------------------------------------------------
// xcat = concat([fa (C), flow (2), bilinear_warp_border(fb, flow) (C)])
// ---------------------------------------------------------------------------
__global__ void warp_concat_kernel(
    const float* __restrict__ fa, const float* __restrict__ fb,
    const float* __restrict__ flow, float* __restrict__ xcat,
    int C, int H, int W)
{
    int HW = H * W;
    int total = (2 * C + 2) * HW;
    int idx = blockIdx.x * blockDim.x + threadIdx.x;
    if (idx >= total) return;
    int c = idx / HW, p = idx - c * HW;
    float v;
    if (c < C) {
        v = fa[c * HW + p];
    } else if (c < C + 2) {
        v = flow[(c - C) * HW + p];
    } else {
        int cc = c - C - 2;
        int y = p / W, x = p - y * W;
        float px = fminf(fmaxf((float)x + flow[p], 0.f), (float)(W - 1));
        float py = fminf(fmaxf((float)y + flow[HW + p], 0.f), (float)(H - 1));
        int x0 = (int)floorf(px), y0 = (int)floorf(py);
        int x1 = min(x0 + 1, W - 1), y1 = min(y0 + 1, H - 1);
        float wx = px - (float)x0, wy = py - (float)y0;
        const float* img = fb + cc * HW;
        v = img[y0 * W + x0] * (1.f - wx) * (1.f - wy)
          + img[y0 * W + x1] * wx * (1.f - wy)
          + img[y1 * W + x0] * (1.f - wx) * wy
          + img[y1 * W + x1] * wx * wy;
    }
    xcat[idx] = v;
}

// ---------------------------------------------------------------------------
// 3x3 conv, pad 1. op: 0 = GELU store, 1 = accumulate into out (flow update).
// ---------------------------------------------------------------------------
__global__ void conv3x3_kernel(
    const float* __restrict__ in, const float* __restrict__ w,
    const float* __restrict__ b, float* __restrict__ out,
    int Cin, int Cout, int H, int W, int op)
{
    int HW = H * W;
    int idx = blockIdx.x * blockDim.x + threadIdx.x;
    if (idx >= Cout * HW) return;
    int co = idx / HW, p = idx - co * HW;
    int y = p / W, x = p - y * W;
    float s = b[co];
    for (int ci = 0; ci < Cin; ++ci) {
        const float* ip = in + ci * HW;
        const float* wp = w + (co * Cin + ci) * 9;
        #pragma unroll
        for (int ky = 0; ky < 3; ++ky) {
            int yy = y + ky - 1;
            if (yy < 0 || yy >= H) continue;
            #pragma unroll
            for (int kx = 0; kx < 3; ++kx) {
                int xx = x + kx - 1;
                if (xx < 0 || xx >= W) continue;
                s += ip[yy * W + xx] * wp[ky * 3 + kx];
            }
        }
    }
    if (op == 0) {
        out[idx] = 0.5f * s * (1.f + erff(s * 0.70710678118654752f));
    } else {
        out[idx] += s;
    }
}

// Bilinear 2x upsample, align_corners=True
__global__ void upsample2x_kernel(const float* __restrict__ in, float* __restrict__ out,
                                  int C, int H, int W)
{
    int Ho = 2 * H, Wo = 2 * W;
    int idx = blockIdx.x * blockDim.x + threadIdx.x;
    if (idx >= C * Ho * Wo) return;
    int c = idx / (Ho * Wo), p = idx - c * Ho * Wo;
    int y = p / Wo, x = p - y * Wo;
    float sx = (float)x * (float)(W - 1) / (float)(Wo - 1);
    float sy = (float)y * (float)(H - 1) / (float)(Ho - 1);
    int x0 = (int)floorf(sx), y0 = (int)floorf(sy);
    int x1 = min(x0 + 1, W - 1), y1 = min(y0 + 1, H - 1);
    float wx = sx - (float)x0, wy = sy - (float)y0;
    const float* img = in + c * H * W;
    out[idx] = img[y0 * W + x0] * (1.f - wx) * (1.f - wy)
             + img[y0 * W + x1] * wx * (1.f - wy)
             + img[y1 * W + x0] * (1.f - wx) * wy
             + img[y1 * W + x1] * wx * wy;
}

// ---------------------------------------------------------------------------
// Host orchestration
// ---------------------------------------------------------------------------
struct TBP {
    const float *ln1w, *ln1b, *inw, *inb, *ow, *ob, *ln2w, *ln2b, *w1, *b1, *w2, *b2;
};

static void run_tblock(float* x, const TBP& p, float* LN, float* QKV, float* ATTN,
                       float* Hb, hipStream_t s)
{
    layernorm_kernel<<<NTOK, 192, 0, s>>>(x, p.ln1w, p.ln1b, LN);
    gemm_wmma_kernel<<<dim3(NTOK / 32, 18), 128, 0, s>>>(LN, p.inw, p.inb, nullptr, QKV,
                                                         NTOK, 192, 576, 0);
    attn_kernel<<<dim3(NTOK / 16, NHEAD), 32, 0, s>>>(QKV, ATTN, NTOK);
    gemm_wmma_kernel<<<dim3(NTOK / 32, 6), 128, 0, s>>>(ATTN, p.ow, p.ob, x, x,
                                                        NTOK, 192, 192, 2);
    layernorm_kernel<<<NTOK, 192, 0, s>>>(x, p.ln2w, p.ln2b, LN);
    gemm_wmma_kernel<<<dim3(NTOK / 32, 24), 128, 0, s>>>(LN, p.w1, p.b1, nullptr, Hb,
                                                         NTOK, 192, 768, 1);
    gemm_wmma_kernel<<<dim3(NTOK / 32, 6), 128, 0, s>>>(Hb, p.w2, p.b2, x, x,
                                                        NTOK, 768, 192, 2);
}

extern "C" void kernel_launch(void* const* d_in, const int* in_sizes, int n_in,
                              void* d_out, int out_size, void* d_ws, size_t ws_size,
                              hipStream_t stream)
{
    (void)in_sizes; (void)n_in; (void)out_size; (void)ws_size;

    const float* feats_l1 = (const float*)d_in[0];  // (3,1,64,64,64)
    const float* feats_l2 = (const float*)d_in[1];  // (3,1,128,32,32)
    const float* tok_w1   = (const float*)d_in[3] + 192 * 49;  // level-1 slice
    const float* tok_b1   = (const float*)d_in[4] + 192;

    auto lcm = [&](int slot) { return (const float*)d_in[5 + slot]; };
    auto gtr = [&](int slot) { return (const float*)d_in[17 + slot]; };
    const float* head_w1 = (const float*)d_in[29];
    const float* head_b1 = (const float*)d_in[30];
    const float* head_w2 = (const float*)d_in[31];
    const float* head_b2 = (const float*)d_in[32];
    const float* ref1_w1 = (const float*)d_in[33];
    const float* ref1_b1 = (const float*)d_in[34];
    const float* ref1_w2 = (const float*)d_in[35];
    const float* ref1_b2 = (const float*)d_in[36];
    const float* ref0_w1 = (const float*)d_in[37];
    const float* ref0_b1 = (const float*)d_in[38];
    const float* ref0_w2 = (const float*)d_in[39];
    const float* ref0_b2 = (const float*)d_in[40];

    char* ws = (char*)d_ws;
    float* X     = (float*)(ws + 0x000000);  // (1024,192)
    float* X0    = (float*)(ws + 0x0C0000);
    float* GX    = (float*)(ws + 0x180000);
    float* LN    = (float*)(ws + 0x240000);
    float* QKV   = (float*)(ws + 0x300000);  // (1024,576)
    float* ATTN  = (float*)(ws + 0x540000);
    float* Hb    = (float*)(ws + 0x600000);  // (1024,768)
    float* HT    = (float*)(ws + 0x900000);
    float* FV    = (float*)(ws + 0x9C0000);  // (1024,2)
    float* C0    = (float*)(ws + 0x9C4000);  // (2,32,32)
    float* C1    = (float*)(ws + 0x9C8000);
    float* XCAT1 = (float*)(ws + 0x9CC000);  // (258,32,32)
    float* MID1  = (float*)(ws + 0xAD0000);  // (128,32,32)
    float* XCAT0 = (float*)(ws + 0xB50000);  // (130,64,64)
    float* MID0  = (float*)(ws + 0xD58000);  // (64,64,64)

    const int XB = NTOK * DMODEL * sizeof(float);
    const int CHW2 = 128 * 1024;  // feats_l2 frame stride
    const int CHW1 = 64 * 4096;   // feats_l1 frame stride

    TBP lp[6], gp[2];
    for (int i = 0; i < 6; ++i) {
        lp[i] = { lcm(0) + i * 192, lcm(1) + i * 192,
                  lcm(2) + i * 576 * 192, lcm(3) + i * 576,
                  lcm(4) + i * 192 * 192, lcm(5) + i * 192,
                  lcm(6) + i * 192, lcm(7) + i * 192,
                  lcm(8) + i * 768 * 192, lcm(9) + i * 768,
                  lcm(10) + i * 192 * 768, lcm(11) + i * 192 };
    }
    for (int i = 0; i < 2; ++i) {
        gp[i] = { gtr(0) + i * 192, gtr(1) + i * 192,
                  gtr(2) + i * 576 * 192, gtr(3) + i * 576,
                  gtr(4) + i * 192 * 192, gtr(5) + i * 192,
                  gtr(6) + i * 192, gtr(7) + i * 192,
                  gtr(8) + i * 768 * 192, gtr(9) + i * 768,
                  gtr(10) + i * 192 * 768, gtr(11) + i * 192 };
    }

    // ---- Tokenize + LCM, pair 0 (level 1 only; other levels are dead code) ----
    corr_tok_kernel<<<NTOK, 192, 0, stream>>>(feats_l2, feats_l2 + CHW2, tok_w1, tok_b1, X);
    for (int i = 0; i < 6; ++i) run_tblock(X, lp[i], LN, QKV, ATTN, Hb, stream);
    (void)hipMemcpyAsync(X0, X, XB, hipMemcpyDeviceToDevice, stream);

    // ---- pair 1 + EMA blend with pair-0 state ----
    corr_tok_kernel<<<NTOK, 192, 0, stream>>>(feats_l2 + CHW2, feats_l2 + 2 * CHW2,
                                              tok_w1, tok_b1, X);
    for (int i = 0; i < 6; ++i) run_tblock(X, lp[i], LN, QKV, ATTN, Hb, stream);
    blend_kernel<<<(NTOK * DMODEL + 255) / 256, 256, 0, stream>>>(X, X0, NTOK * DMODEL);

    // ---- GTR + head -> coarse flows ----
    float* coarse[2] = { C0, C1 };
    float* src[2] = { X0, X };
    for (int k = 0; k < 2; ++k) {
        (void)hipMemcpyAsync(GX, src[k], XB, hipMemcpyDeviceToDevice, stream);
        for (int i = 0; i < 2; ++i) run_tblock(GX, gp[i], LN, QKV, ATTN, Hb, stream);
        gemm_wmma_kernel<<<dim3(NTOK / 32, 6), 128, 0, stream>>>(GX, head_w1, head_b1,
                                                                 nullptr, HT, NTOK, 192, 192, 1);
        gemm_wmma_kernel<<<dim3(NTOK / 32, 1), 128, 0, stream>>>(HT, head_w2, head_b2,
                                                                 nullptr, FV, NTOK, 192, 2, 0);
        nc_to_cn_kernel<<<(2 * NTOK + 255) / 256, 256, 0, stream>>>(FV, coarse[k], NTOK);
    }

    // ---- Recurrent decoder ----
    for (int k = 0; k < 2; ++k) {
        const float* fa1 = feats_l2 + k * CHW2;
        const float* fb1 = feats_l2 + (k + 1) * CHW2;
        const float* fa0 = feats_l1 + k * CHW1;
        const float* fb0 = feats_l1 + (k + 1) * CHW1;
        float* cur = coarse[k];
        for (int it = 0; it < 4; ++it) {
            warp_concat_kernel<<<(258 * 1024 + 255) / 256, 256, 0, stream>>>(
                fa1, fb1, cur, XCAT1, 128, 32, 32);
            conv3x3_kernel<<<(128 * 1024 + 255) / 256, 256, 0, stream>>>(
                XCAT1, ref1_w1, ref1_b1, MID1, 258, 128, 32, 32, 0);
            conv3x3_kernel<<<(2 * 1024 + 255) / 256, 256, 0, stream>>>(
                MID1, ref1_w2, ref1_b2, cur, 128, 2, 32, 32, 1);
        }
        float* cur2 = (float*)d_out + k * 2 * 64 * 64;  // level-0 flow lives in d_out
        upsample2x_kernel<<<(2 * 64 * 64 + 255) / 256, 256, 0, stream>>>(cur, cur2, 2, 32, 32);
        for (int it = 0; it < 4; ++it) {
            warp_concat_kernel<<<(130 * 4096 + 255) / 256, 256, 0, stream>>>(
                fa0, fb0, cur2, XCAT0, 64, 64, 64);
            conv3x3_kernel<<<(64 * 4096 + 255) / 256, 256, 0, stream>>>(
                XCAT0, ref0_w1, ref0_b1, MID0, 130, 64, 64, 64, 0);
            conv3x3_kernel<<<(2 * 4096 + 255) / 256, 256, 0, stream>>>(
                MID0, ref0_w2, ref0_b2, cur2, 64, 2, 64, 64, 1);
        }
    }
}
